// network_50027779064061
// MI455X (gfx1250) — matrix operand order repaired and checked by reading, not compiled, not run on hardware
//
#include <hip/hip_runtime.h>

// D-MPNN on MI455X: bf16 edge-state storage (halves HBM traffic at 23.3 TB/s),
// v_wmma_f32_16x16x32_bf16 for all dense GEMMs, f32 atomic segment sums.

#define N_NODES  400000
#define N_EDGES  1600000
#define N_GRAPHS 10000
#define HID      64
#define SLOPE    0.01f

typedef __bf16 bhalf;
typedef __attribute__((ext_vector_type(16))) __bf16 v16bf;
typedef __attribute__((ext_vector_type(8)))  __bf16 v8bf;
typedef __attribute__((ext_vector_type(4)))  __bf16 v4bf;
typedef __attribute__((ext_vector_type(8)))  float  v8f;
typedef __attribute__((ext_vector_type(4)))  float  v4f;

__device__ __forceinline__ float lrelu(float x) { return x > 0.f ? x : SLOPE * x; }

__device__ __forceinline__ v8f zero8() {
  v8f z;
#pragma unroll
  for (int i = 0; i < 8; i++) z[i] = 0.f;
  return z;
}

__device__ __forceinline__ v8f wmma_bf16(v16bf a, v16bf b, v8f c) {
  return __builtin_amdgcn_wmma_f32_16x16x32_bf16(false, a, false, b, (short)0, c, false, false);
}

// B fragment (32x16 K-major) from transposed weights WT[n][Kpad] in LDS.
// Lanes 0-15 hold K=0..7,16..23 (col N = lane), lanes 16-31 hold K=8..15,24..31.
__device__ __forceinline__ v16bf load_bfrag(const bhalf* WT, int Kpad, int t, int c, int lane) {
  const bhalf* wrow = WT + (t * 16 + (lane & 15)) * Kpad + c * 32 + ((lane & 16) ? 8 : 0);
  v8bf lo = *(const v8bf*)(wrow);
  v8bf hi = *(const v8bf*)(wrow + 16);
  v16bf r;
#pragma unroll
  for (int i = 0; i < 8; i++) { r[i] = lo[i]; r[8 + i] = hi[i]; }
  return r;
}

// A fragment (16x32 bf16) K-chunk c from a bf16 row (row M = lane&15).
// elems 0..7 -> K = c*32 + (lane<16?0:8) + i ; elems 8..15 -> c*32 + (lane<16?16:24) + i
__device__ __forceinline__ v16bf load_afrag(const bhalf* row, int c, int lane) {
  int b0 = c * 32 + ((lane & 16) ? 8 : 0);
  int b1 = c * 32 + ((lane & 16) ? 24 : 16);
  v8bf lo = *(const v8bf*)(row + b0);
  v8bf hi = *(const v8bf*)(row + b1);
  v16bf r;
#pragma unroll
  for (int i = 0; i < 8; i++) { r[i] = lo[i]; r[8 + i] = hi[i]; }
  return r;
}

// ---------------- weight prep: transpose to [n][Kpad] bf16, fuse edge branch -----------
__global__ void __launch_bounds__(256) k_prep(
    const float* __restrict__ Wn, const float* __restrict__ We, const float* __restrict__ Wi,
    const float* __restrict__ Wa, const float* __restrict__ Wl,
    const float* __restrict__ be, const float* __restrict__ bi,
    bhalf* __restrict__ WnT, bhalf* __restrict__ WiTopT, bhalf* __restrict__ WfT,
    float* __restrict__ biasf, bhalf* __restrict__ WaT, bhalf* __restrict__ WlT) {
  int tid = threadIdx.x;
  for (int idx = tid; idx < 64 * 96; idx += 256) {           // Wn^T, K pad 74->96
    int n = idx / 96, k = idx % 96;
    WnT[idx] = (bhalf)((k < 74) ? Wn[k * 64 + n] : 0.f);
  }
  for (int idx = tid; idx < 64 * 64; idx += 256) {           // Wi top half (node part)
    int n = idx / 64, k = idx % 64;
    WiTopT[idx] = (bhalf)Wi[k * 64 + n];
  }
  for (int idx = tid; idx < 64 * 32; idx += 256) {           // Wfused = We @ Wi_bot, K pad 13->32
    int n = idx / 32, k = idx % 32;
    float v = 0.f;
    if (k < 13)
      for (int j = 0; j < 64; j++) v += We[k * 64 + j] * Wi[(64 + j) * 64 + n];
    WfT[idx] = (bhalf)v;
  }
  for (int n = tid; n < 64; n += 256) {                      // bias' = be @ Wi_bot + bi
    float v = bi[n];
    for (int j = 0; j < 64; j++) v += be[j] * Wi[(64 + j) * 64 + n];
    biasf[n] = v;
  }
  for (int idx = tid; idx < 64 * 128; idx += 256) {          // Wa^T
    int n = idx / 128, k = idx % 128;
    WaT[idx] = (bhalf)Wa[k * 64 + n];
  }
  for (int idx = tid; idx < 4 * 64 * 64; idx += 256) {       // Wl^T per layer
    int l = idx / 4096, r = idx % 4096;
    int n = r / 64, k = r % 64;
    WlT[idx] = (bhalf)Wl[l * 4096 + k * 64 + n];
  }
}

// ---------------- node = node_feats @ Wn + bn   (linear, store bf16) --------------------
__global__ void __launch_bounds__(256) k_node_embed(
    const float* __restrict__ nf, const float* __restrict__ bn,
    const bhalf* __restrict__ WnT, bhalf* __restrict__ node_bf) {
  __shared__ __align__(16) bhalf sW[64 * 96];
  __shared__ float sb[64];
  __shared__ __align__(16) bhalf stage[8][1024];
  for (int i = threadIdx.x; i < 64 * 96; i += 256) sW[i] = WnT[i];
  if (threadIdx.x < 64) sb[threadIdx.x] = bn[threadIdx.x];
  __syncthreads();
  const int lane = threadIdx.x & 31, wave = threadIdx.x >> 5;
  const int ntiles = N_NODES / 16;
  for (int tile = blockIdx.x * 8 + wave; tile < ntiles; tile += gridDim.x * 8) {
    const int row = tile * 16 + (lane & 15);
    const float* ar = nf + (size_t)row * 74;
    v16bf a[3];
#pragma unroll
    for (int c = 0; c < 3; c++) {
      int b0 = c * 32 + ((lane & 16) ? 8 : 0), b1 = c * 32 + ((lane & 16) ? 24 : 16);
#pragma unroll
      for (int i = 0; i < 8; i++) {
        int k0 = b0 + i, k1 = b1 + i;
        a[c][i]     = (bhalf)((k0 < 74) ? ar[k0] : 0.f);
        a[c][8 + i] = (bhalf)((k1 < 74) ? ar[k1] : 0.f);
      }
    }
    v8f acc[4];
#pragma unroll
    for (int t = 0; t < 4; t++) {
      acc[t] = zero8();
#pragma unroll
      for (int c = 0; c < 3; c++) acc[t] = wmma_bf16(a[c], load_bfrag(sW, 96, t, c, lane), acc[t]);
    }
    bhalf* st = stage[wave];
    const int M0 = (lane & 16) ? 8 : 0, cl = lane & 15;
#pragma unroll
    for (int t = 0; t < 4; t++) {
      float b = sb[t * 16 + cl];
#pragma unroll
      for (int r = 0; r < 8; r++) st[(M0 + r) * 64 + t * 16 + cl] = (bhalf)(acc[t][r] + b);
    }
    v8bf* gout = (v8bf*)(node_bf + (size_t)tile * 1024);
    const v8bf* sv = (const v8bf*)st;
#pragma unroll
    for (int i = 0; i < 4; i++) gout[lane + 32 * i] = sv[lane + 32 * i];
  }
}

// ---------------- h0 = lrelu(node[src]@Wi_top + edge_feats@Wfused + bias') --------------
__global__ void __launch_bounds__(256) k_edge_init(
    const bhalf* __restrict__ node_bf, const float* __restrict__ ef,
    const int* __restrict__ src,
    const bhalf* __restrict__ WiTopT, const bhalf* __restrict__ WfT,
    const float* __restrict__ biasf, bhalf* __restrict__ h0) {
  __shared__ __align__(16) bhalf sWt[64 * 64];
  __shared__ __align__(16) bhalf sWf[64 * 32];
  __shared__ float sb[64];
  __shared__ __align__(16) bhalf stage[8][1024];
  for (int i = threadIdx.x; i < 64 * 64; i += 256) sWt[i] = WiTopT[i];
  for (int i = threadIdx.x; i < 64 * 32; i += 256) sWf[i] = WfT[i];
  if (threadIdx.x < 64) sb[threadIdx.x] = biasf[threadIdx.x];
  __syncthreads();
  const int lane = threadIdx.x & 31, wave = threadIdx.x >> 5;
  const int ntiles = N_EDGES / 16;
  for (int tile = blockIdx.x * 8 + wave; tile < ntiles; tile += gridDim.x * 8) {
    const int row = tile * 16 + (lane & 15);
    const int s = src[row];
    const bhalf* nr = node_bf + (size_t)s * 64;
    v16bf a0 = load_afrag(nr, 0, lane);
    v16bf a1 = load_afrag(nr, 1, lane);
    const float* er = ef + (size_t)row * 13;
    v16bf ae;
    {
      int b0 = (lane & 16) ? 8 : 0, b1 = (lane & 16) ? 24 : 16;
#pragma unroll
      for (int i = 0; i < 8; i++) {
        int k0 = b0 + i, k1 = b1 + i;
        ae[i]     = (bhalf)((k0 < 13) ? er[k0] : 0.f);
        ae[8 + i] = (bhalf)((k1 < 13) ? er[k1] : 0.f);
      }
    }
    v8f acc[4];
#pragma unroll
    for (int t = 0; t < 4; t++) {
      acc[t] = zero8();
      acc[t] = wmma_bf16(a0, load_bfrag(sWt, 64, t, 0, lane), acc[t]);
      acc[t] = wmma_bf16(a1, load_bfrag(sWt, 64, t, 1, lane), acc[t]);
      acc[t] = wmma_bf16(ae, load_bfrag(sWf, 32, t, 0, lane), acc[t]);
    }
    bhalf* st = stage[wave];
    const int M0 = (lane & 16) ? 8 : 0, cl = lane & 15;
#pragma unroll
    for (int t = 0; t < 4; t++) {
      float b = sb[t * 16 + cl];
#pragma unroll
      for (int r = 0; r < 8; r++) st[(M0 + r) * 64 + t * 16 + cl] = (bhalf)lrelu(acc[t][r] + b);
    }
    v8bf* gout = (v8bf*)(h0 + (size_t)tile * 1024);
    const v8bf* sv = (const v8bf*)st;
#pragma unroll
    for (int i = 0; i < 4; i++) gout[lane + 32 * i] = sv[lane + 32 * i];
  }
}

// ---------------- zero f32 buffer --------------------------------------------------------
__global__ void __launch_bounds__(256) k_zero(float* __restrict__ p, long long n) {
  long long stride = (long long)gridDim.x * blockDim.x;
  for (long long i = (long long)blockIdx.x * blockDim.x + threadIdx.x; i < n; i += stride)
    p[i] = 0.f;
}

// ---------------- out[index[r]] += rows[r]  (bf16 rows -> f32 atomic accumulate) --------
__global__ void __launch_bounds__(256) k_scatter(
    const bhalf* __restrict__ rows, const int* __restrict__ index,
    float* __restrict__ out, int nrows) {
  long long total = (long long)nrows * 16;
  long long stride = (long long)gridDim.x * blockDim.x;
  for (long long idx = (long long)blockIdx.x * blockDim.x + threadIdx.x; idx < total; idx += stride) {
    long long e = idx >> 4;
    int g = (int)(idx & 15);
    int d = index[e];
    v4bf hv = *(const v4bf*)(rows + e * 64 + g * 4);
    float* ap = out + (size_t)d * 64 + g * 4;
#pragma unroll
    for (int j = 0; j < 4; j++)
      __hip_atomic_fetch_add(ap + j, (float)hv[j], __ATOMIC_RELAXED, __HIP_MEMORY_SCOPE_AGENT);
  }
}

// ---------------- h = lrelu(h0 + (agg[src] - h[rev]) @ Wl + bl)  (in-place safe) --------
__global__ void __launch_bounds__(256) k_conv(
    const bhalf* __restrict__ h_in, bhalf* __restrict__ h_out,
    const bhalf* __restrict__ h0, const float* __restrict__ agg,
    const int* __restrict__ src,
    const bhalf* __restrict__ WlT, const float* __restrict__ bl) {
  __shared__ __align__(16) bhalf sW[64 * 64];
  __shared__ float sb[64];
  __shared__ __align__(16) bhalf stage[8][1024];
  for (int i = threadIdx.x; i < 64 * 64; i += 256) sW[i] = WlT[i];
  if (threadIdx.x < 64) sb[threadIdx.x] = bl[threadIdx.x];
  __syncthreads();
  const int lane = threadIdx.x & 31, wave = threadIdx.x >> 5;
  const int ntiles = N_EDGES / 16;
  for (int tile = blockIdx.x * 8 + wave; tile < ntiles; tile += gridDim.x * 8) {
    const int row = tile * 16 + (lane & 15);
    const int s = src[row];
    const float* ag = agg + (size_t)s * 64;
    const bhalf* hr = h_in + (size_t)(row ^ 1) * 64;   // rev edge lives in same tile
    v16bf a[2];
#pragma unroll
    for (int c = 0; c < 2; c++) {
      int b0 = c * 32 + ((lane & 16) ? 8 : 0), b1 = c * 32 + ((lane & 16) ? 24 : 16);
      v4f g0 = *(const v4f*)(ag + b0), g1 = *(const v4f*)(ag + b0 + 4);
      v4f g2 = *(const v4f*)(ag + b1), g3 = *(const v4f*)(ag + b1 + 4);
      v8bf r0 = *(const v8bf*)(hr + b0), r1 = *(const v8bf*)(hr + b1);
#pragma unroll
      for (int i = 0; i < 4; i++) {
        a[c][i]      = (bhalf)(g0[i] - (float)r0[i]);
        a[c][4 + i]  = (bhalf)(g1[i] - (float)r0[4 + i]);
        a[c][8 + i]  = (bhalf)(g2[i] - (float)r1[i]);
        a[c][12 + i] = (bhalf)(g3[i] - (float)r1[4 + i]);
      }
    }
    bhalf* st = stage[wave];
    {  // stage h0 tile (contiguous b128 copy)
      const v8bf* h0v = (const v8bf*)(h0 + (size_t)tile * 1024);
      v8bf* sv = (v8bf*)st;
#pragma unroll
      for (int i = 0; i < 4; i++) sv[lane + 32 * i] = h0v[lane + 32 * i];
    }
    v8f acc[4];
#pragma unroll
    for (int t = 0; t < 4; t++) {
      acc[t] = zero8();
      acc[t] = wmma_bf16(a[0], load_bfrag(sW, 64, t, 0, lane), acc[t]);
      acc[t] = wmma_bf16(a[1], load_bfrag(sW, 64, t, 1, lane), acc[t]);
    }
    const int M0 = (lane & 16) ? 8 : 0, cl = lane & 15;
#pragma unroll
    for (int t = 0; t < 4; t++) {
      float b = sb[t * 16 + cl];
#pragma unroll
      for (int r = 0; r < 8; r++) {
        int o = (M0 + r) * 64 + t * 16 + cl;
        float v = lrelu(acc[t][r] + b + (float)st[o]);
        st[o] = (bhalf)v;
      }
    }
    v8bf* gout = (v8bf*)(h_out + (size_t)tile * 1024);
    const v8bf* sv = (const v8bf*)st;
#pragma unroll
    for (int i = 0; i < 4; i++) gout[lane + 32 * i] = sv[lane + 32 * i];
  }
}

// ---------------- node_h = lrelu(concat(node, agg) @ Wa + ba) ---------------------------
__global__ void __launch_bounds__(256) k_readout(
    const bhalf* __restrict__ node_bf, const float* __restrict__ agg,
    const float* __restrict__ ba, const bhalf* __restrict__ WaT,
    bhalf* __restrict__ node_h) {
  __shared__ __align__(16) bhalf sW[64 * 128];
  __shared__ float sb[64];
  __shared__ __align__(16) bhalf stage[8][1024];
  for (int i = threadIdx.x; i < 64 * 128; i += 256) sW[i] = WaT[i];
  if (threadIdx.x < 64) sb[threadIdx.x] = ba[threadIdx.x];
  __syncthreads();
  const int lane = threadIdx.x & 31, wave = threadIdx.x >> 5;
  const int ntiles = N_NODES / 16;
  for (int tile = blockIdx.x * 8 + wave; tile < ntiles; tile += gridDim.x * 8) {
    const int row = tile * 16 + (lane & 15);
    const bhalf* nr = node_bf + (size_t)row * 64;
    const float* ar = agg + (size_t)row * 64;
    v16bf a[4];
    a[0] = load_afrag(nr, 0, lane);
    a[1] = load_afrag(nr, 1, lane);
#pragma unroll
    for (int c = 0; c < 2; c++) {
      int b0 = c * 32 + ((lane & 16) ? 8 : 0), b1 = c * 32 + ((lane & 16) ? 24 : 16);
      v4f g0 = *(const v4f*)(ar + b0), g1 = *(const v4f*)(ar + b0 + 4);
      v4f g2 = *(const v4f*)(ar + b1), g3 = *(const v4f*)(ar + b1 + 4);
#pragma unroll
      for (int i = 0; i < 4; i++) {
        a[2 + c][i]      = (bhalf)g0[i];
        a[2 + c][4 + i]  = (bhalf)g1[i];
        a[2 + c][8 + i]  = (bhalf)g2[i];
        a[2 + c][12 + i] = (bhalf)g3[i];
      }
    }
    v8f acc[4];
#pragma unroll
    for (int t = 0; t < 4; t++) {
      acc[t] = zero8();
#pragma unroll
      for (int c = 0; c < 4; c++) acc[t] = wmma_bf16(a[c], load_bfrag(sW, 128, t, c, lane), acc[t]);
    }
    bhalf* st = stage[wave];
    const int M0 = (lane & 16) ? 8 : 0, cl = lane & 15;
#pragma unroll
    for (int t = 0; t < 4; t++) {
      float b = sb[t * 16 + cl];
#pragma unroll
      for (int r = 0; r < 8; r++) st[(M0 + r) * 64 + t * 16 + cl] = (bhalf)lrelu(acc[t][r] + b);
    }
    v8bf* gout = (v8bf*)(node_h + (size_t)tile * 1024);
    const v8bf* sv = (const v8bf*)st;
#pragma unroll
    for (int i = 0; i < 4; i++) gout[lane + 32 * i] = sv[lane + 32 * i];
  }
}

// ---------------- head: out = lrelu(g@W1+b1) @ W2 + b2 ----------------------------------
__global__ void __launch_bounds__(256) k_head(
    const float* __restrict__ g, const float* __restrict__ W1,
    const float* __restrict__ b1, const float* __restrict__ W2,
    const float* __restrict__ b2, float* __restrict__ out) {
  __shared__ float sW1[64 * 64];
  __shared__ float sb1[64];
  __shared__ float sW2[64];
  for (int i = threadIdx.x; i < 4096; i += 256) sW1[i] = W1[i];
  if (threadIdx.x < 64) { sb1[threadIdx.x] = b1[threadIdx.x]; sW2[threadIdx.x] = W2[threadIdx.x]; }
  __syncthreads();
  int gi = blockIdx.x * 256 + threadIdx.x;
  if (gi >= N_GRAPHS) return;
  const float* gr = g + (size_t)gi * 64;
  float r[64];
#pragma unroll
  for (int k = 0; k < 64; k++) r[k] = gr[k];
  float o = b2[0];
  for (int j = 0; j < 64; j++) {
    float acc = sb1[j];
#pragma unroll
    for (int k = 0; k < 64; k++) acc += r[k] * sW1[k * 64 + j];
    o += lrelu(acc) * sW2[j];
  }
  out[gi] = o;
}

extern "C" void kernel_launch(void* const* d_in, const int* in_sizes, int n_in,
                              void* d_out, int out_size, void* d_ws, size_t ws_size,
                              hipStream_t stream) {
  (void)in_sizes; (void)n_in; (void)out_size; (void)ws_size;
  const float* node_feats = (const float*)d_in[0];
  const float* edge_feats = (const float*)d_in[1];
  const int*   src        = (const int*)d_in[2];
  const int*   dst        = (const int*)d_in[3];
  const int*   node_gid   = (const int*)d_in[4];
  const float* Wn = (const float*)d_in[5];
  const float* bn = (const float*)d_in[6];
  const float* We = (const float*)d_in[7];
  const float* be = (const float*)d_in[8];
  const float* Wi = (const float*)d_in[9];
  const float* bi = (const float*)d_in[10];
  const float* Wa = (const float*)d_in[11];
  const float* ba = (const float*)d_in[12];
  const float* Wl = (const float*)d_in[13];
  const float* bl = (const float*)d_in[14];
  const float* W1 = (const float*)d_in[15];
  const float* b1 = (const float*)d_in[16];
  const float* W2 = (const float*)d_in[17];
  const float* b2 = (const float*)d_in[18];

  char* ws = (char*)d_ws;
  size_t off = 0;
  auto alloc = [&](size_t bytes) -> void* {
    void* p = ws + off;
    off = (off + bytes + 255) & ~(size_t)255;
    return p;
  };
  bhalf* WnT    = (bhalf*)alloc(64 * 96 * 2);
  bhalf* WiTopT = (bhalf*)alloc(64 * 64 * 2);
  bhalf* WfT    = (bhalf*)alloc(64 * 32 * 2);
  float* biasf  = (float*)alloc(64 * 4);
  bhalf* WaT    = (bhalf*)alloc(64 * 128 * 2);
  bhalf* WlT    = (bhalf*)alloc(4 * 64 * 64 * 2);
  bhalf* node_bf = (bhalf*)alloc((size_t)N_NODES * 64 * 2);
  bhalf* h0      = (bhalf*)alloc((size_t)N_EDGES * 64 * 2);
  bhalf* h       = (bhalf*)alloc((size_t)N_EDGES * 64 * 2);
  float* agg     = (float*)alloc((size_t)N_NODES * 64 * 4);
  float* g       = (float*)alloc((size_t)N_GRAPHS * 64 * 4);
  bhalf* node_h  = h0;  // h0 is dead after the last conv layer; reuse

  k_prep<<<1, 256, 0, stream>>>(Wn, We, Wi, Wa, Wl, be, bi,
                                WnT, WiTopT, WfT, biasf, WaT, WlT);
  k_node_embed<<<1024, 256, 0, stream>>>(node_feats, bn, WnT, node_bf);
  k_edge_init<<<2048, 256, 0, stream>>>(node_bf, edge_feats, src, WiTopT, WfT, biasf, h0);

  const bhalf* hin = h0;
  for (int l = 0; l < 4; l++) {
    k_zero<<<2048, 256, 0, stream>>>(agg, (long long)N_NODES * 64);
    k_scatter<<<4096, 256, 0, stream>>>(hin, dst, agg, N_EDGES);
    k_conv<<<2048, 256, 0, stream>>>(hin, h, h0, agg, src, WlT + l * 4096, bl + l * 64);
    hin = h;
  }

  k_zero<<<2048, 256, 0, stream>>>(agg, (long long)N_NODES * 64);
  k_scatter<<<4096, 256, 0, stream>>>(h, dst, agg, N_EDGES);
  k_readout<<<1024, 256, 0, stream>>>(node_bf, agg, ba, WaT, node_h);

  k_zero<<<64, 256, 0, stream>>>(g, (long long)N_GRAPHS * 64);
  k_scatter<<<1024, 256, 0, stream>>>(node_h, node_gid, g, N_NODES);
  k_head<<<(N_GRAPHS + 255) / 256, 256, 0, stream>>>(g, W1, b1, W2, b2, (float*)d_out);
}
